// Transformer_64647847739728
// MI455X (gfx1250) — matrix operand order repaired and checked by reading, not compiled, not run on hardware
//
#include <hip/hip_runtime.h>

typedef __attribute__((ext_vector_type(16))) __bf16 v16bf;
typedef __attribute__((ext_vector_type(8)))  __bf16 v8bf;
typedef __attribute__((ext_vector_type(8)))  float  v8f;
typedef __attribute__((ext_vector_type(4)))  unsigned int v4u;
typedef __attribute__((ext_vector_type(8)))  int v8i;
typedef __attribute__((ext_vector_type(4)))  int v4i;

namespace {
constexpr int kD   = 1024;   // d_model
constexpr int kFF  = 4096;   // ffn dim
constexpr int kV   = 32000;  // vocab
constexpr int kB   = 8;      // batch
constexpr int kS   = 512;    // seq
constexpr int kL   = 6;      // layers (weight-shared)

constexpr int BM  = 128;
constexpr int BN  = 64;
constexpr int BKt = 32;
constexpr int ATile = BM * BKt;   // elements per A LDS buffer
constexpr int BTile = BN * BKt;   // elements per B LDS buffer
}

#if defined(__has_builtin)
#  if __has_builtin(__builtin_amdgcn_tensor_load_to_lds) && __has_builtin(__builtin_amdgcn_s_wait_tensorcnt)
#    define USE_TDM 1
#  endif
#endif
#ifndef USE_TDM
#  define USE_TDM 0
#endif

#if USE_TDM
#  define TDM_WAIT(n) __builtin_amdgcn_s_wait_tensorcnt(n)
#else
#  define TDM_WAIT(n) ((void)0)
#endif

// fp32 -> bf16 round-to-nearest-even
__device__ __forceinline__ __bf16 f2bf(float f) {
  unsigned u = __builtin_bit_cast(unsigned, f);
  unsigned r = u + 0x7FFFu + ((u >> 16) & 1u);
  unsigned short h = (unsigned short)(r >> 16);
  return __builtin_bit_cast(__bf16, h);
}

// ---- Tensor Data Mover wrappers (D# per CDNA5 ISA) ----
#if USE_TDM
__device__ __forceinline__ void tdm_load(v4u g0, v8i g1, v4i g2, v4i g3) {
#if __clang_major__ >= 23
  v8i g4 = {};
  __builtin_amdgcn_tensor_load_to_lds(g0, g1, g2, g3, g4, 0);
#else
  __builtin_amdgcn_tensor_load_to_lds(g0, g1, g2, g3, 0);
#endif
}
#else
__device__ __forceinline__ void tdm_load(v4u, v8i, v4i, v4i) {}
#endif

// D# group1 for a 2D bf16 tile: tile_dim0 = BKt (contiguous k), tile_dim1 = rows,
// tensor_dim0/1 = safe upper bounds (no OOB: all dims divide tiles exactly).
__device__ __forceinline__ v8i tdm_group1(int tensd0, int tensd1, int rows, int ld) {
  v8i g = {};
  g[0] = (int)(1u << 16);                                         // data_size=2B
  g[1] = (int)(((unsigned)tensd0 & 0xFFFFu) << 16);               // tensor_dim0 lo16
  g[2] = (int)((((unsigned)tensd0 >> 16) & 0xFFFFu) |
               (((unsigned)tensd1 & 0xFFFFu) << 16));             // dim0 hi | dim1 lo
  g[3] = (int)((((unsigned)tensd1 >> 16) & 0xFFFFu) |
               ((unsigned)BKt << 16));                            // dim1 hi | tile_dim0
  g[4] = (int)((unsigned)rows & 0xFFFFu);                         // tile_dim1 (tile_dim2=0)
  g[5] = ld;                                                      // tensor_dim0_stride lo32
  g[6] = 0; g[7] = 0;
  return g;
}
__device__ __forceinline__ v4u tdm_group0(unsigned lds_addr, unsigned long gaddr) {
  v4u g;
  g[0] = 1u;                                            // count=1 (valid descriptor)
  g[1] = lds_addr;                                      // LDS byte address
  g[2] = (unsigned)gaddr;                               // global addr lo32
  g[3] = (unsigned)((gaddr >> 32) & 0x01FFFFFFu) | (2u << 30);  // hi bits | type=2
  return g;
}

// C[M,N] = act( scale * (A ? B) + bias ), bf16 A/B operands, f32 accumulate.
// TA==0: A [M,K] row-major.  TA==1: A [K,M] (C = A^T ...).
// TB==1: B [N,K] row-major (C = A @ B^T).  TB==0: B [K,N] (C = A @ B).
// OBF==1: store bf16, else f32. strA/strB/strC: per-batch strides (blockIdx.z).
template<int TA, int TB, int SILU, int OBF>
__global__ __launch_bounds__(256)
void gemm_bf16_wmma(const __bf16* __restrict__ A, const __bf16* __restrict__ Bm,
                    const float* __restrict__ bias, void* __restrict__ Cv,
                    int M, int N, int K, int lda, int ldb, int ldc,
                    long strA, long strB, long strC, float scale)
{
  // double-buffered for the TDM pipeline (24 KB total; WGP has 320 KB)
  __shared__ alignas(16) __bf16 sA[2 * ATile];   // [buf][m][k]
  __shared__ alignas(16) __bf16 sB[2 * BTile];   // [buf][n][k]

  const int tid = threadIdx.x;
  A  += strA * blockIdx.z;
  Bm += strB * blockIdx.z;
  const int m0 = blockIdx.y * BM;
  const int n0 = blockIdx.x * BN;

  const int lane = tid & 31;
  const int wave = tid >> 5;   // 0..7
  const int wm   = wave >> 1;  // 0..3
  const int wn   = wave & 1;   // 0..1
  const int half = lane >> 4;  // 0/1
  const int l16  = lane & 15;

  constexpr bool kTDM = USE_TDM && (TA == 0) && (TB == 1);

  v8f acc[2][2] = {};

  // fragment-load + 4x WMMA on one LDS buffer
  auto compute_tile = [&](const __bf16* tA, const __bf16* tB) {
    // A 16x32 frag: lanes 0-15 K=0..7 & 16..23; lanes 16-31 K=8..15 & 24..31.
    // B 32x16 frag: lanes 0-15 K=0..7 & 16..23; lanes 16-31 K=8..15 & 24..31 (col = lane%16).
    v16bf af[2], bfm[2];
    #pragma unroll
    for (int i = 0; i < 2; ++i) {
      const int m = wm * 32 + i * 16 + l16;
      v8bf lo = *reinterpret_cast<const v8bf*>(&tA[m * BKt + half * 8]);
      v8bf hi = *reinterpret_cast<const v8bf*>(&tA[m * BKt + 16 + half * 8]);
      af[i] = __builtin_shufflevector(lo, hi, 0,1,2,3,4,5,6,7,8,9,10,11,12,13,14,15);
    }
    #pragma unroll
    for (int j = 0; j < 2; ++j) {
      const int n = wn * 32 + j * 16 + l16;
      v8bf lo = *reinterpret_cast<const v8bf*>(&tB[n * BKt + half * 8]);
      v8bf hi = *reinterpret_cast<const v8bf*>(&tB[n * BKt + 16 + half * 8]);
      bfm[j] = __builtin_shufflevector(lo, hi, 0,1,2,3,4,5,6,7,8,9,10,11,12,13,14,15);
    }
    #pragma unroll
    for (int i = 0; i < 2; ++i)
      #pragma unroll
      for (int j = 0; j < 2; ++j)
        acc[i][j] = __builtin_amdgcn_wmma_f32_16x16x32_bf16(
            false, af[i], false, bfm[j], (short)0, acc[i][j], false, false);
  };

  if constexpr (kTDM) {
    // ---- software-pipelined tensor-DMA double buffering ----
    const v8i g1A = tdm_group1(K, M, BM, lda);
    const v8i g1B = tdm_group1(K, N, BN, ldb);
    const v4i z4 = {};
    const unsigned ldsA = (unsigned)(unsigned long)(void*)sA;  // low 32 = LDS offset
    const unsigned ldsB = (unsigned)(unsigned long)(void*)sB;
    const unsigned long gA0 = (unsigned long)A + (((long)m0 * lda) << 1);
    const unsigned long gB0 = (unsigned long)Bm + (((long)n0 * ldb) << 1);
    const int ntiles = K / BKt;

    if (wave == 0) {  // prologue: tile 0 -> buffer 0 (TDM ignores EXEC; 1 DMA/wave)
      tdm_load(tdm_group0(ldsA, gA0), g1A, z4, z4);
      tdm_load(tdm_group0(ldsB, gB0), g1B, z4, z4);
    }
    for (int t = 0; t < ntiles; ++t) {
      const int cur = t & 1;
      if (wave == 0) {
        if (t + 1 < ntiles) {  // prefetch tile t+1 into the other buffer
          const unsigned long koff = ((unsigned long)(t + 1) * BKt) << 1;
          tdm_load(tdm_group0(ldsA + (unsigned)((cur ^ 1) * ATile * 2), gA0 + koff),
                   g1A, z4, z4);
          tdm_load(tdm_group0(ldsB + (unsigned)((cur ^ 1) * BTile * 2), gB0 + koff),
                   g1B, z4, z4);
          TDM_WAIT(2);   // in-order: <=2 outstanding => tile t's 2 loads landed
        } else {
          TDM_WAIT(0);   // drain for the last tile
        }
      }
      __syncthreads();                 // publish buffer `cur` to all waves
      compute_tile(&sA[cur * ATile], &sB[cur * BTile]);
      __syncthreads();                 // all reads of `cur` done before its reuse
    }
  } else {
    // ---- manual staging (transposed operands / no-TDM fallback) ----
    for (int k0 = 0; k0 < K; k0 += BKt) {
      if constexpr (TA == 0) {
        const int row = tid >> 1;            // 0..127
        const int kh  = (tid & 1) * 16;      // 16 bf16 = 32B
        const uint4* s = reinterpret_cast<const uint4*>(A + (long)(m0 + row) * lda + k0 + kh);
        uint4* d = reinterpret_cast<uint4*>(&sA[row * BKt + kh]);
        d[0] = s[0]; d[1] = s[1];
        if (k0 + BKt < K) __builtin_prefetch(s + 2, 0, 1);
      } else {  // A [K,M]: transpose into LDS
        const int mloc = tid & 127;
        const int kh   = (tid >> 7) * 16;
        #pragma unroll
        for (int j = 0; j < 16; ++j)
          sA[mloc * BKt + kh + j] = A[(long)(k0 + kh + j) * lda + (m0 + mloc)];
      }
      if constexpr (TB == 1) {
        const int n  = tid >> 2;             // 0..63
        const int kh = (tid & 3) * 8;        // 8 bf16 = 16B
        *reinterpret_cast<uint4*>(&sB[n * BKt + kh]) =
            *reinterpret_cast<const uint4*>(Bm + (long)(n0 + n) * ldb + k0 + kh);
      } else {  // B [K,N]: transpose into LDS
        const int n  = tid & 63;
        const int kh = (tid >> 6) * 8;
        #pragma unroll
        for (int j = 0; j < 8; ++j)
          sB[n * BKt + kh + j] = Bm[(long)(k0 + kh + j) * ldb + (n0 + n)];
      }
      __syncthreads();
      compute_tile(sA, sB);
      __syncthreads();
    }
  }

  // epilogue: C/D layout: lane<16 -> M=v, lane>=16 -> M=8+v; N = lane%16
  #pragma unroll
  for (int j = 0; j < 2; ++j) {
    const int n = n0 + wn * 32 + j * 16 + l16;
    const float bv = bias ? bias[n] : 0.0f;
    #pragma unroll
    for (int i = 0; i < 2; ++i) {
      #pragma unroll
      for (int v = 0; v < 8; ++v) {
        const int m = m0 + wm * 32 + i * 16 + half * 8 + v;
        float val = acc[i][j][v] * scale + bv;
        if constexpr (SILU) val = val * (1.0f / (1.0f + __expf(-val)));
        if constexpr (OBF) ((__bf16*)Cv + strC * blockIdx.z)[(long)m * ldc + n] = f2bf(val);
        else               ((float*)Cv  + strC * blockIdx.z)[(long)m * ldc + n] = val;
      }
    }
  }
  (void)M; (void)N;
}

// LayerNorm over D=1024 per row; f32 in, bf16 out (consumers are GEMMs).
__global__ __launch_bounds__(256)
void layernorm_k(const float* __restrict__ x, const float* __restrict__ g,
                 const float* __restrict__ b, __bf16* __restrict__ y)
{
  const int row = blockIdx.x;
  const float* xr = x + (long)row * kD;
  __bf16* yr = y + (long)row * kD;
  float s = 0.f, ss = 0.f;
  for (int i = threadIdx.x; i < kD; i += 256) { float v = xr[i]; s += v; ss += v * v; }
  __shared__ float rs[256], rq[256];
  rs[threadIdx.x] = s; rq[threadIdx.x] = ss;
  __syncthreads();
  for (int off = 128; off > 0; off >>= 1) {
    if (threadIdx.x < off) { rs[threadIdx.x] += rs[threadIdx.x + off]; rq[threadIdx.x] += rq[threadIdx.x + off]; }
    __syncthreads();
  }
  const float mu  = rs[0] * (1.0f / kD);
  const float var = rq[0] * (1.0f / kD) - mu * mu;
  const float inv = rsqrtf(var + 1e-5f);
  for (int i = threadIdx.x; i < kD; i += 256)
    yr[i] = f2bf((xr[i] - mu) * inv * g[i] + b[i]);
}

// out[t][d] = emb[tok[t]][d] + pe(batch(t), d)   (reference PE indexed by batch)
__global__ __launch_bounds__(256)
void embed_pe_k(const int* __restrict__ tok, const float* __restrict__ emb,
                float* __restrict__ out)
{
  const int t = blockIdx.x;
  const int bpos = t >> 9;
  const int id = tok[t];
  const float* e = emb + (long)id * kD;
  float* o = out + (long)t * kD;
  const float kCoef = -0.008994473f;  // -ln(10000)/1024
  for (int d = threadIdx.x; d < kD; d += 256) {
    const int i2 = d & ~1;
    const float ang = (float)bpos * __expf((float)i2 * kCoef);
    const float pe = (d & 1) ? __cosf(ang) : __sinf(ang);
    o[d] = e[d] + pe;
  }
}

// softmax over keys (axis Sk) of scores [B][Sk][Sq]; writes bf16 alpha.
__global__ __launch_bounds__(256)
void softmax_keys_k(const float* __restrict__ sc, __bf16* __restrict__ al)
{
  const int idx = blockIdx.x * 256 + threadIdx.x;  // b*S + q
  const int b = idx >> 9;
  const int q = idx & (kS - 1);
  const float* base = sc + (long)b * kS * kS + q;
  __bf16* abase = al + (long)b * kS * kS + q;
  float mx = -1e30f;
  for (int k = 0; k < kS; ++k) mx = fmaxf(mx, base[(long)k * kS]);
  float sum = 0.f;
  for (int k = 0; k < kS; ++k) sum += __expf(base[(long)k * kS] - mx);
  const float inv = 1.0f / sum;
  for (int k = 0; k < kS; ++k)
    abase[(long)k * kS] = f2bf(__expf(base[(long)k * kS] - mx) * inv);
}

// fp32 -> bf16 bulk convert (n multiple of 4)
__global__ __launch_bounds__(256)
void f32_to_bf16_k(const float* __restrict__ in, __bf16* __restrict__ out, long n4)
{
  const long i = (long)blockIdx.x * 256 + threadIdx.x;
  if (i >= n4) return;
  float4 f = reinterpret_cast<const float4*>(in)[i];
  __bf16* o = out + i * 4;
  o[0] = f2bf(f.x); o[1] = f2bf(f.y); o[2] = f2bf(f.z); o[3] = f2bf(f.w);
}

extern "C" void kernel_launch(void* const* d_in, const int* in_sizes, int n_in,
                              void* d_out, int out_size, void* d_ws, size_t ws_size,
                              hipStream_t stream)
{
  enum {
    I_EMB = 0,
    E_KW, E_KB, E_VW, E_VB, E_QW, E_QB, E_OW, E_OB,
    E_F1W, E_F1B, E_F2W, E_F2B, E_L1G, E_L1B, E_L2G, E_L2B,
    S_KW, S_KB, S_VW, S_VB, S_QW, S_QB, S_OW, S_OB,      // dec.sa
    C_KW, C_KB, C_VW, C_VB, C_QW, C_QB, C_OW, C_OB,      // dec.ca
    D_F1W, D_F1B, D_F2W, D_F2B, D_L1G, D_L1B, D_L2G, D_L2B, D_L3G, D_L3B,
    CLS_W, CLS_B, I_SRC, I_TGT
  };
  auto F = [&](int i) { return (const float*)d_in[i]; };
  const int* src = (const int*)d_in[I_SRC];
  const int* tgt = (const int*)d_in[I_TGT];

  const int  NTOK = kB * kS;               // 4096
  const long AEL  = (long)NTOK * kD;       // [B,S,D]

  // ---- workspace bump allocator ----
  char* p = (char*)d_ws;
  auto allocF = [&](long n) { float* r = (float*)p; p += n * 4; return r; };
  auto allocH = [&](long n) { __bf16* r = (__bf16*)p; p += n * 2; return r; };

  float*  x   = allocF(AEL);                 // encoder acts -> memory (f32)
  float*  y   = allocF(AEL);                 // decoder acts (f32)
  float*  h   = allocF(AEL);                 // f32 temp (attn outputs pre-LN)
  float*  sc  = allocF((long)kB * kS * kS);  // scores f32
  __bf16* t0b = allocH(AEL);                 // LN / converted acts
  __bf16* kb  = allocH(AEL);
  __bf16* vb  = allocH(AEL);
  __bf16* qb  = allocH(AEL);
  __bf16* t4b = allocH(AEL);                 // attention output
  __bf16* kcb = allocH(AEL);                 // hoisted cross K(mem)
  __bf16* vcb = allocH(AEL);                 // hoisted cross V(mem)
  __bf16* ffb = allocH((long)NTOK * kFF);
  __bf16* al  = allocH((long)kB * kS * kS);  // softmax alpha (bf16)

  // ---- pre-convert all GEMM weights to bf16 (reused 6x across shared layers) ----
  const long DDsz = (long)kD * kD, FDsz = (long)kFF * kD, CLsz = (long)kV * kD;
  const int  wIdx[17] = { E_KW,E_VW,E_QW,E_OW,E_F1W,E_F2W,
                          S_KW,S_VW,S_QW,S_OW, C_KW,C_VW,C_QW,C_OW,
                          D_F1W,D_F2W, CLS_W };
  const long wSz[17]  = { DDsz,DDsz,DDsz,DDsz,FDsz,FDsz,
                          DDsz,DDsz,DDsz,DDsz,DDsz,DDsz,DDsz,DDsz,
                          FDsz,FDsz, CLsz };
  const __bf16* WB[47] = {};
  for (int i = 0; i < 17; ++i) {
    __bf16* wb = allocH(wSz[i]);
    const long n4 = wSz[i] / 4;
    f32_to_bf16_k<<<dim3((unsigned)((n4 + 255) / 256)), dim3(256), 0, stream>>>(F(wIdx[i]), wb, n4);
    WB[wIdx[i]] = wb;
  }

  // ---- GEMM helpers ----
  auto LINb = [&](const __bf16* A, int wi, int bi, __bf16* C,
                  int M, int N, int K, bool silu) {   // bf16 out
    dim3 g(N / BN, M / BM, 1), blk(256);
    if (silu)
      gemm_bf16_wmma<0,1,1,1><<<g, blk, 0, stream>>>(A, WB[wi], F(bi), C,
          M, N, K, K, K, N, 0, 0, 0, 1.0f);
    else
      gemm_bf16_wmma<0,1,0,1><<<g, blk, 0, stream>>>(A, WB[wi], F(bi), C,
          M, N, K, K, K, N, 0, 0, 0, 1.0f);
  };
  auto LINf = [&](const __bf16* A, int wi, int bi, float* C,
                  int M, int N, int K) {              // f32 out
    dim3 g(N / BN, M / BM, 1), blk(256);
    gemm_bf16_wmma<0,1,0,0><<<g, blk, 0, stream>>>(A, WB[wi], F(bi), C,
        M, N, K, K, K, N, 0, 0, 0, 1.0f);
  };
  auto LNORM = [&](const float* in, __bf16* out, int gi, int bi) {
    layernorm_k<<<NTOK, 256, 0, stream>>>(in, F(gi), F(bi), out);
  };
  // scores = K·Q^T/sqrt(D) -> softmax over keys -> out = alpha^T · V
  auto ATT_CORE = [&](const __bf16* kB_, const __bf16* vB_, const __bf16* qB_, __bf16* out) {
    gemm_bf16_wmma<0,1,0,0><<<dim3(kS / BN, kS / BM, kB), dim3(256), 0, stream>>>(
        kB_, qB_, nullptr, sc, kS, kS, kD, kD, kD, kS,
        (long)kS * kD, (long)kS * kD, (long)kS * kS, 0.03125f);
    softmax_keys_k<<<NTOK / 256, 256, 0, stream>>>(sc, al);
    gemm_bf16_wmma<1,0,0,1><<<dim3(kD / BN, kS / BM, kB), dim3(256), 0, stream>>>(
        al, vB_, nullptr, out, kS, kD, kS, kS, kD, kD,
        (long)kS * kS, (long)kS * kD, (long)kS * kD, 1.0f);
  };
  auto CVT = [&](const float* in, __bf16* out, long n) {
    const long n4 = n / 4;
    f32_to_bf16_k<<<dim3((unsigned)((n4 + 255) / 256)), dim3(256), 0, stream>>>(in, out, n4);
  };

  embed_pe_k<<<NTOK, 256, 0, stream>>>(src, F(I_EMB), x);
  embed_pe_k<<<NTOK, 256, 0, stream>>>(tgt, F(I_EMB), y);

  // ---- encoder (weight-shared, applied kL times; no residuals per reference) ----
  for (int l = 0; l < kL; ++l) {
    LNORM(x, t0b, E_L1G, E_L1B);
    LINb(t0b, E_KW, E_KB, kb, NTOK, kD, kD, false);
    LINb(t0b, E_VW, E_VB, vb, NTOK, kD, kD, false);
    LINb(t0b, E_QW, E_QB, qb, NTOK, kD, kD, false);
    ATT_CORE(kb, vb, qb, t4b);
    LINf(t4b, E_OW, E_OB, x, NTOK, kD, kD);
    LNORM(x, t0b, E_L2G, E_L2B);
    LINb(t0b, E_F1W, E_F1B, ffb, NTOK, kFF, kD, true);   // SiLU fused
    LINf(ffb, E_F2W, E_F2B, x, NTOK, kD, kFF);
  }

  // ---- hoist cross-attn K/V of mem (same weights & mem every decoder layer) ----
  CVT(x, t0b, AEL);
  LINb(t0b, C_KW, C_KB, kcb, NTOK, kD, kD, false);
  LINb(t0b, C_VW, C_VB, vcb, NTOK, kD, kD, false);

  // ---- decoder ----
  for (int l = 0; l < kL; ++l) {
    LNORM(y, t0b, D_L1G, D_L1B);
    LINb(t0b, S_KW, S_KB, kb, NTOK, kD, kD, false);
    LINb(t0b, S_VW, S_VB, vb, NTOK, kD, kD, false);
    LINb(t0b, S_QW, S_QB, qb, NTOK, kD, kD, false);
    ATT_CORE(kb, vb, qb, t4b);
    LINf(t4b, S_OW, S_OB, h, NTOK, kD, kD);              // y1
    LNORM(h, t0b, D_L2G, D_L2B);
    LINb(t0b, C_QW, C_QB, qb, NTOK, kD, kD, false);      // cross Q
    ATT_CORE(kcb, vcb, qb, t4b);
    LINf(t4b, C_OW, C_OB, h, NTOK, kD, kD);              // y2
    LNORM(h, t0b, D_L3G, D_L3B);
    LINb(t0b, D_F1W, D_F1B, ffb, NTOK, kFF, kD, true);
    LINf(ffb, D_F2W, D_F2B, y, NTOK, kD, kFF);
  }

  // ---- classifier -> d_out [B,S,V] f32 ----
  CVT(y, t0b, AEL);
  LINf(t0b, CLS_W, CLS_B, (float*)d_out, NTOK, kV, kD);

  (void)in_sizes; (void)n_in; (void)out_size; (void)ws_size;
}
